// BaseHiViT_26345329394207
// MI455X (gfx1250) — matrix operand order repaired and checked by reading, not compiled, not run on hardware
//
#include <hip/hip_runtime.h>
#include <math.h>

// ---------------------------------------------------------------------------
// HiViT-Base forward for MI455X (gfx1250, wave32, WMMA).
// All GEMMs run through v_wmma_f32_16x16x32_bf16 (bf16 inputs, f32 accum).
// f32 weights are converted to bf16 while staging into LDS (L2-resident).
// ---------------------------------------------------------------------------

typedef __attribute__((ext_vector_type(16))) __bf16 v16bf;
typedef __attribute__((ext_vector_type(8)))  float  v8f;

#define TBM 128
#define TBN 128
#define TBK 32
#define KP  40   // LDS row stride in halfs (80B, 16B-aligned, de-conflicted)

// ---------------------------------------------------------------------------
// Fused WMMA GEMM:  C = epilogue( A[M,K](bf16) * B^T + bias )
//  BL   : 0 -> B is [N,K] row-major (weights / attention K-matrix)
//         1 -> B is [K,N] row-major (attention V), transposed via LDS
//  BF32 : 1 -> B is f32 (convert to bf16 on the way into LDS)
//  ACT  : 1 -> exact GELU
//  OUTK : 0 -> f32 out, 1 -> bf16 out
//  RES  : 1 -> add f32 residual (same layout as C)
// Batched via gridDim.z: z -> (b = z/zdiv, h = z%zdiv) pointer strides.
// ---------------------------------------------------------------------------
template<int BL, int BF32, int ACT, int OUTK, int RES>
__global__ __launch_bounds__(256)
void hivit_gemm(const __bf16* __restrict__ A, const void* __restrict__ Bv,
                const float* __restrict__ bias, const float* __restrict__ resid,
                void* __restrict__ Cv,
                int M, int N, int K, int ldA, int ldB, int ldC, int zdiv,
                long aSB, long aSH, long bSB, long bSH, long cSB, long cSH)
{
    __shared__ __bf16 lsA[TBM * KP];
    __shared__ __bf16 lsB[TBN * KP];

    const int tid  = threadIdx.x;
    const int wid  = tid >> 5;
    const int lane = tid & 31;
    const int wm   = (wid & 1) * 64;   // wave M offset in block tile
    const int wn   = (wid >> 1) * 32;  // wave N offset in block tile
    const int blockM = blockIdx.y * TBM;
    const int blockN = blockIdx.x * TBN;
    const int z  = blockIdx.z;
    const int zb = z / zdiv, zh = z % zdiv;

    const __bf16* Ab = A + (long)zb * aSB + (long)zh * aSH;
    const long bOff  = (long)zb * bSB + (long)zh * bSH;
    const long cOff  = (long)zb * cSB + (long)zh * cSH;

    const int  lh   = lane & 15;
    const bool hiH  = lane >= 16;
    const int  kloA = hiH ? 16 : 0;   // A k order used here (contiguous): lanes 16-31 hold K 16..31
    const int  kloB = hiH ? 16 : 0;   // B k order used here (contiguous): lanes 16-31 hold K 16..31

    v8f acc[4][2];
#pragma unroll
    for (int i = 0; i < 4; ++i)
#pragma unroll
        for (int j = 0; j < 2; ++j) acc[i][j] = {};

    for (int kt = 0; kt < K; kt += TBK) {
        // ---- stage A tile (128 x 32 bf16) into LDS ----
#pragma unroll
        for (int it = 0; it < 2; ++it) {
            int u = tid + it * 256;              // 512 uint4 slots
            int r = u >> 2;
            int q = (u & 3) * 8;
            int gRow = blockM + r; if (gRow >= M) gRow = M - 1;
            const uint4* src = reinterpret_cast<const uint4*>(
                Ab + (long)gRow * ldA + kt + q);
            *reinterpret_cast<uint4*>(&lsA[r * KP + q]) = *src;
        }
        // ---- stage B tile into LDS (as [N][K] bf16) ----
        if constexpr (BL == 0) {
#pragma unroll
            for (int it = 0; it < 2; ++it) {
                int u = tid + it * 256;
                int r = u >> 2;
                int q = (u & 3) * 8;
                int gN = blockN + r; if (gN >= N) gN = N - 1;
                if constexpr (BF32) {
                    const float* bp = (const float*)Bv + bOff + (long)gN * ldB + kt + q;
                    float4 f0 = *reinterpret_cast<const float4*>(bp);
                    float4 f1 = *reinterpret_cast<const float4*>(bp + 4);
                    union { uint4 u4; __bf16 e[8]; } cv;
                    cv.e[0] = (__bf16)f0.x; cv.e[1] = (__bf16)f0.y;
                    cv.e[2] = (__bf16)f0.z; cv.e[3] = (__bf16)f0.w;
                    cv.e[4] = (__bf16)f1.x; cv.e[5] = (__bf16)f1.y;
                    cv.e[6] = (__bf16)f1.z; cv.e[7] = (__bf16)f1.w;
                    *reinterpret_cast<uint4*>(&lsB[r * KP + q]) = cv.u4;
                } else {
                    const uint4* bp = reinterpret_cast<const uint4*>(
                        (const __bf16*)Bv + bOff + (long)gN * ldB + kt + q);
                    *reinterpret_cast<uint4*>(&lsB[r * KP + q]) = *bp;
                }
            }
        } else {
            // B is [K][N] bf16: read N-contiguous, transpose into LDS [N][K]
            int kk = tid >> 3;                   // 0..31 (K row in tile)
            int nn = (tid & 7) * 16;             // 0..112 (N col in tile)
            union { uint4 u4[2]; __bf16 e[16]; } tb;
            const uint4* bp = reinterpret_cast<const uint4*>(
                (const __bf16*)Bv + bOff + (long)(kt + kk) * ldB + blockN + nn);
            tb.u4[0] = bp[0]; tb.u4[1] = bp[1];
#pragma unroll
            for (int i = 0; i < 16; ++i)
                lsB[(nn + i) * KP + kk] = tb.e[i];
        }
        __syncthreads();

        // ---- build fragments and issue 8 WMMAs ----
        union Frag { uint4 u4[2]; v16bf v; };
        Frag af[4], bfg[2];
#pragma unroll
        for (int i = 0; i < 4; ++i) {
            int base = (wm + i * 16 + lh) * KP;
            af[i].u4[0] = *reinterpret_cast<const uint4*>(&lsA[base + kloA]);
            af[i].u4[1] = *reinterpret_cast<const uint4*>(&lsA[base + kloA + 8]);
        }
#pragma unroll
        for (int j = 0; j < 2; ++j) {
            int base = (wn + j * 16 + lh) * KP;
            bfg[j].u4[0] = *reinterpret_cast<const uint4*>(&lsB[base + kloB]);
            bfg[j].u4[1] = *reinterpret_cast<const uint4*>(&lsB[base + kloB + 8]);
        }
#pragma unroll
        for (int i = 0; i < 4; ++i)
#pragma unroll
            for (int j = 0; j < 2; ++j)
                acc[i][j] = __builtin_amdgcn_wmma_f32_16x16x32_bf16(
                    false, af[i].v, false, bfg[j].v, (short)0, acc[i][j],
                    false, false);
        __syncthreads();
    }

    // ---- epilogue: bias / residual / GELU / store (guarded) ----
#pragma unroll
    for (int i = 0; i < 4; ++i) {
#pragma unroll
        for (int j = 0; j < 2; ++j) {
            int gcol = blockN + wn + j * 16 + lh;
            if (gcol >= N) continue;
            float bv = bias ? bias[gcol] : 0.f;
#pragma unroll
            for (int r = 0; r < 8; ++r) {
                int grow = blockM + wm + i * 16 + (hiH ? 8 : 0) + r;
                if (grow >= M) continue;
                long o = cOff + (long)grow * ldC + gcol;
                float v = acc[i][j][r] + bv;
                if constexpr (RES) v += resid[o];
                if constexpr (ACT) v = 0.5f * v * (1.0f + erff(v * 0.70710678118f));
                if constexpr (OUTK == 0) ((float*)Cv)[o] = v;
                else                     ((__bf16*)Cv)[o] = (__bf16)v;
            }
        }
    }
}

// --------------------------- LayerNorm (wave/row) ---------------------------
__global__ __launch_bounds__(256)
void ln_kernel(const float* __restrict__ x, const float* __restrict__ w,
               const float* __restrict__ b, void* __restrict__ out,
               int M, int D, int outf32)
{
    int wid = threadIdx.x >> 5, lane = threadIdx.x & 31;
    int row = blockIdx.x * 8 + wid;
    if (row >= M) return;
    const float* xr = x + (long)row * D;
    float s = 0.f, ss = 0.f;
    for (int c = lane; c < D; c += 32) { float v = xr[c]; s += v; ss += v * v; }
#pragma unroll
    for (int off = 16; off; off >>= 1) {
        s  += __shfl_xor(s,  off, 32);
        ss += __shfl_xor(ss, off, 32);
    }
    float mean = s / D;
    float inv  = rsqrtf(ss / D - mean * mean + 1e-5f);
    for (int c = lane; c < D; c += 32) {
        float v = (xr[c] - mean) * inv * w[c] + b[c];
        if (outf32) ((float*)out)[(long)row * D + c] = v;
        else        ((__bf16*)out)[(long)row * D + c] = (__bf16)v;
    }
}

// -------------------- patch-merge gather + LayerNorm ------------------------
__global__ __launch_bounds__(256)
void merge_ln_kernel(const float* __restrict__ y, const float* __restrict__ w,
                     const float* __restrict__ b, __bf16* __restrict__ out,
                     int M2, int Sin, int D)
{
    int wid = threadIdx.x >> 5, lane = threadIdx.x & 31;
    int r2 = blockIdx.x * 8 + wid;
    if (r2 >= M2) return;
    int So = Sin >> 1;
    int token = r2 / (So * So);
    int s  = r2 % (So * So);
    int i2 = s / So, j2 = s % So;
    const int di[4] = {0, 1, 0, 1}, dj[4] = {0, 0, 1, 1};
    int D4 = 4 * D;
    int nt = D4 >> 5;                       // per-lane elements (<= 32)
    float vb[32];
    float sum = 0.f, ssum = 0.f;
    for (int t = 0; t < nt; ++t) {
        int ct = lane + t * 32;
        int g  = ct / D;
        int srow = token * Sin * Sin + (2 * i2 + di[g]) * Sin + (2 * j2 + dj[g]);
        float v = y[(long)srow * D + (ct - g * D)];
        vb[t] = v; sum += v; ssum += v * v;
    }
#pragma unroll
    for (int off = 16; off; off >>= 1) {
        sum  += __shfl_xor(sum,  off, 32);
        ssum += __shfl_xor(ssum, off, 32);
    }
    float mean = sum / D4;
    float inv  = rsqrtf(ssum / D4 - mean * mean + 1e-5f);
    for (int t = 0; t < nt; ++t) {
        int ct = lane + t * 32;
        out[(long)r2 * D4 + ct] = (__bf16)((vb[t] - mean) * inv * w[ct] + b[ct]);
    }
}

// ------------------- patch-embed conv (4x4 s4, 3->128) ----------------------
__global__ __launch_bounds__(128)
void conv_kernel(const float* __restrict__ x, const float* __restrict__ w,
                 const float* __restrict__ bias, float* __restrict__ out)
{
    __shared__ float xs[48];
    int r = blockIdx.x;                     // token-inner row, 0..50175
    int bI = r / 3136, rem = r % 3136;
    int p = rem >> 4, inner = rem & 15;
    int hp = p / 14, wp = p % 14;
    int H  = hp * 4 + (inner >> 2);
    int W  = wp * 4 + (inner & 3);
    int t = threadIdx.x;
    if (t < 48) {
        int cin = t / 16, kk = t % 16;
        int kh = kk / 4, kw = kk % 4;
        xs[t] = x[((long)(bI * 3 + cin) * 224 + (H * 4 + kh)) * 224 + (W * 4 + kw)];
    }
    __syncthreads();
    int c = t;
    float sum = bias[c];
    const float* wc = w + (long)c * 48;
#pragma unroll 8
    for (int k = 0; k < 48; ++k) sum += xs[k] * wc[k];
    out[(long)r * 128 + c] = sum;
}

// -------------------------- pos-embed add -----------------------------------
__global__ void pos_add_kernel(float* __restrict__ y, const float* __restrict__ pos)
{
    long i = (long)blockIdx.x * blockDim.x + threadIdx.x;
    if (i >= 3136L * 512) return;
    long row = i / 512, col = i % 512;
    y[i] += pos[(row % 196) * 512 + col];
}

// --------------- softmax over keys with analytic RPE bias -------------------
__global__ __launch_bounds__(256)
void softmax_rpe_kernel(const float* __restrict__ scores,
                        const float* __restrict__ rpe,
                        __bf16* __restrict__ probs)
{
    __shared__ float red[256];
    int q = blockIdx.x;          // 0..195
    int z = blockIdx.y;          // 0..127 (b*8+h)
    int h = z & 7;
    int k = threadIdx.x;
    int qi = q / 14, qj = q % 14;
    float logit = -1e30f;
    if (k < 196) {
        int ki = k / 14, kj = k % 14;
        int idx = (qi - ki + 13) * 27 + (qj - kj + 13);
        logit = scores[((long)z * 196 + q) * 224 + k] * 0.125f + rpe[idx * 8 + h];
    }
    red[k] = logit; __syncthreads();
    for (int s = 128; s; s >>= 1) {
        if (k < s) red[k] = fmaxf(red[k], red[k + s]);
        __syncthreads();
    }
    float mx = red[0]; __syncthreads();
    float e = (k < 196) ? __expf(logit - mx) : 0.f;
    red[k] = e; __syncthreads();
    for (int s = 128; s; s >>= 1) {
        if (k < s) red[k] += red[k + s];
        __syncthreads();
    }
    float inv = 1.f / red[0];
    if (k < 224)
        probs[((long)z * 196 + q) * 224 + k] = (__bf16)((k < 196) ? e * inv : 0.f);
}

// --------------------------- final copy -------------------------------------
__global__ void copy_kernel(const float* __restrict__ src, float* __restrict__ dst, long n)
{
    long i = (long)blockIdx.x * blockDim.x + threadIdx.x;
    if (i < n) dst[i] = src[i];
}

// ---------------------------------------------------------------------------
static inline int cdiv_i(int a, int b) { return (a + b - 1) / b; }

template<int BL, int BF32, int ACT, int OUTK, int RES>
static void rungemm(const __bf16* A, const void* B, const float* bias,
                    const float* resid, void* C,
                    int M, int N, int K, int ldA, int ldB, int ldC,
                    int batches, int zdiv,
                    long aSB, long aSH, long bSB, long bSH, long cSB, long cSH,
                    hipStream_t st)
{
    dim3 g((unsigned)cdiv_i(N, TBN), (unsigned)cdiv_i(M, TBM), (unsigned)batches);
    hivit_gemm<BL, BF32, ACT, OUTK, RES><<<g, 256, 0, st>>>(
        A, B, bias, resid, C, M, N, K, ldA, ldB, ldC, zdiv,
        aSB, aSH, bSB, bSH, cSB, cSH);
}

extern "C" void kernel_launch(void* const* d_in, const int* in_sizes, int n_in,
                              void* d_out, int out_size, void* d_ws, size_t ws_size,
                              hipStream_t stream)
{
    (void)in_sizes; (void)n_in; (void)ws_size;
    int ii = 0;
    auto F = [&]() { return (const float*)d_in[ii++]; };

    const float* x      = F();
    const float* conv_w = F();  const float* conv_b = F();
    const float* pn_w   = F();  const float* pn_b   = F();
    const float* pos    = F();  const float* rpe    = F();
    struct MB { const float *nw, *nb, *f1w, *f1b, *f2w, *f2b; };
    MB stem[2][4];
    for (int s = 0; s < 2; ++s)
        for (int k = 0; k < 4; ++k)
            stem[s][k] = { F(), F(), F(), F(), F(), F() };
    struct MG { const float *nw, *nb, *rw; };
    MG mg[2];
    for (int m = 0; m < 2; ++m) mg[m] = { F(), F(), F() };
    struct TB { const float *n1w, *n1b, *qw, *qb, *pw, *pb, *n2w, *n2b,
                            *f1w, *f1b, *f2w, *f2b; };
    TB mn[10];
    for (int t = 0; t < 10; ++t)
        mn[t] = { F(), F(), F(), F(), F(), F(), F(), F(), F(), F(), F(), F() };

    // ---- workspace layout (bytes, 256-aligned) ----
    char* ws = (char*)d_ws;
    auto al = [](size_t v) { return (v + 255) & ~(size_t)255; };
    size_t offY   = 0;
    size_t offXN  = al(offY  + (size_t)50176 * 128 * 4);   // y   : f32 residual
    size_t offH   = al(offXN + (size_t)50176 * 128 * 2);   // xn  : bf16 ln-out / attn-out
    size_t offQKV = al(offH  + (size_t)50176 * 384 * 2);   // h   : bf16 hidden / scores+probs
    float*  y    = (float*)(ws + offY);
    __bf16* xn   = (__bf16*)(ws + offXN);
    __bf16* hbuf = (__bf16*)(ws + offH);
    __bf16* qkv  = (__bf16*)(ws + offQKV);                 // (3136+64) x 1536
    float*  ctmp   = (float*)(ws + offH);                  // conv raw (alias h)
    float*  scores = (float*)(ws + offH);                  // 128 x 196 x 224 f32
    __bf16* probs  = (__bf16*)(ws + offH + (size_t)128 * 196 * 224 * 4);

    // ---- patch embed: conv -> LN (f32 residual stream) ----
    conv_kernel<<<50176, 128, 0, stream>>>(x, conv_w, conv_b, ctmp);
    ln_kernel<<<cdiv_i(50176, 8), 256, 0, stream>>>(ctmp, pn_w, pn_b, y, 50176, 128, 1);

    // ---- stem stages: 4 MLP blocks each, then patch merge ----
    const int dims[2] = {128, 256}, hid[2] = {384, 768}, Ms[2] = {50176, 12544};
    for (int st = 0; st < 2; ++st) {
        int D = dims[st], Hd = hid[st], M = Ms[st];
        for (int bk = 0; bk < 4; ++bk) {
            const MB& B = stem[st][bk];
            ln_kernel<<<cdiv_i(M, 8), 256, 0, stream>>>(y, B.nw, B.nb, xn, M, D, 0);
            rungemm<0, 1, 1, 1, 0>(xn, B.f1w, B.f1b, nullptr, hbuf,
                                   M, Hd, D, D, D, Hd, 1, 1, 0, 0, 0, 0, 0, 0, stream);
            rungemm<0, 1, 0, 0, 1>(hbuf, B.f2w, B.f2b, y, y,
                                   M, D, Hd, Hd, Hd, D, 1, 1, 0, 0, 0, 0, 0, 0, stream);
        }
        int M2 = M / 4, Sin = (st == 0) ? 4 : 2;
        merge_ln_kernel<<<cdiv_i(M2, 8), 256, 0, stream>>>(y, mg[st].nw, mg[st].nb,
                                                           xn, M2, Sin, D);
        rungemm<0, 1, 0, 0, 0>(xn, mg[st].rw, nullptr, nullptr, y,
                               M2, 2 * D, 4 * D, 4 * D, 4 * D, 2 * D,
                               1, 1, 0, 0, 0, 0, 0, 0, stream);
    }

    // ---- absolute position embedding ----
    pos_add_kernel<<<cdiv_i(3136 * 512, 256), 256, 0, stream>>>(y, pos);

    // ---- main transformer blocks ----
    for (int t = 0; t < 10; ++t) {
        const TB& B = mn[t];
        // attention
        ln_kernel<<<cdiv_i(3136, 8), 256, 0, stream>>>(y, B.n1w, B.n1b, xn, 3136, 512, 0);
        rungemm<0, 1, 0, 1, 0>(xn, B.qw, B.qb, nullptr, qkv,
                               3136, 1536, 512, 512, 512, 1536,
                               1, 1, 0, 0, 0, 0, 0, 0, stream);
        // scores[bh] = Q @ K^T   (128 batched 196x196x64)
        rungemm<0, 0, 0, 0, 0>(qkv, qkv + 512, nullptr, nullptr, scores,
                               196, 196, 64, 1536, 1536, 224, 128, 8,
                               196L * 1536, 64, 196L * 1536, 64,
                               8L * 196 * 224, 196L * 224, stream);
        softmax_rpe_kernel<<<dim3(196, 128), 256, 0, stream>>>(scores, rpe, probs);
        // attn_out[bh] = P @ V   (K padded to 224; pad probs are zero)
        rungemm<1, 0, 0, 1, 0>(probs, qkv + 1024, nullptr, nullptr, xn,
                               196, 64, 224, 224, 1536, 512, 128, 8,
                               8L * 196 * 224, 196L * 224, 196L * 1536, 64,
                               196L * 512, 64, stream);
        rungemm<0, 1, 0, 0, 1>(xn, B.pw, B.pb, y, y,
                               3136, 512, 512, 512, 512, 512,
                               1, 1, 0, 0, 0, 0, 0, 0, stream);
        // MLP
        ln_kernel<<<cdiv_i(3136, 8), 256, 0, stream>>>(y, B.n2w, B.n2b, xn, 3136, 512, 0);
        rungemm<0, 1, 1, 1, 0>(xn, B.f1w, B.f1b, nullptr, hbuf,
                               3136, 2048, 512, 512, 512, 2048,
                               1, 1, 0, 0, 0, 0, 0, 0, stream);
        rungemm<0, 1, 0, 0, 1>(hbuf, B.f2w, B.f2b, y, y,
                               3136, 512, 2048, 2048, 2048, 512,
                               1, 1, 0, 0, 0, 0, 0, 0, stream);
    }

    copy_kernel<<<cdiv_i(out_size, 256), 256, 0, stream>>>(y, (float*)d_out, out_size);
}
